// EncoderBlock_90409061581196
// MI455X (gfx1250) — matrix operand order repaired and checked by reading, not compiled, not run on hardware
//
#include <hip/hip_runtime.h>
#include <math.h>

// ---------------------------------------------------------------------------
// MI455X / gfx1250 encoder block. All GEMM-like ops use v_wmma_f32_16x16x32_bf16
// (bf16 A/B, f32 accumulate). wave32, 128-thread blocks (4 waves).
// GEMM tile: 64(M) x 128(N), K-step 32, double-buffered LDS, 1 barrier/step,
// each wave owns 32x64 = 2x4 WMMA tiles (8 WMMA per wave per K step).
// ---------------------------------------------------------------------------

typedef __bf16 bf16_t;
typedef __attribute__((ext_vector_type(16))) __bf16 v16bf;
typedef __attribute__((ext_vector_type(8)))  __bf16 v8bf;
typedef __attribute__((ext_vector_type(8)))  float  v8f;

#define B_  32
#define S_  512
#define C_  256
#define H_  8
#define NC_ 4
#define KW_ 7

__device__ __forceinline__ v16bf frag_cat(v8bf lo, v8bf hi) {
  return __builtin_shufflevector(lo, hi, 0,1,2,3,4,5,6,7,8,9,10,11,12,13,14,15);
}

// ---------------------------------------------------------------------------
// Generic batched bf16 WMMA GEMM:  C = alpha * A @ (B or B^T) [+bias][+res][relu]
//   A: M x K row-major (lda), B: K x N row-major (NN) or N x K row-major (NT)
//   grid = (N/128, M/64, batch), block = 128 threads.
//   NN path requires ldb % 8 == 0 (true for all uses here: ldb = 256).
// ---------------------------------------------------------------------------
template<bool BT, bool OBF16, bool RELU, bool HAS_BIAS, bool HAS_RES>
__global__ __launch_bounds__(128)
void gemm_bf16_wmma(const bf16_t* __restrict__ Ag, size_t strideA, int lda,
                    const bf16_t* __restrict__ Bg, size_t strideB, int ldb,
                    void* __restrict__ Cv, size_t strideC, int ldc,
                    const float* __restrict__ bias,
                    const float* __restrict__ res, size_t strideR,
                    float alpha, int M, int N, int K)
{
  __shared__ __align__(16) bf16_t Als[2][64][40];    // [buf][m][k], pitch 80B
  __shared__ __align__(16) bf16_t Bls[2][128][40];   // [buf][n][k]
  const int t    = threadIdx.x;
  const int lane = t & 31;
  const int wave = t >> 5;
  const int half = lane >> 4;
  const int l16  = lane & 15;
  const int wm   = (wave >> 1) << 5;   // 0 / 32
  const int wn   = (wave & 1) << 6;    // 0 / 64
  const int m0   = blockIdx.y * 64;
  const int n0   = blockIdx.x * 128;
  const size_t bz = blockIdx.z;
  const bf16_t* A = Ag + bz * strideA;
  const bf16_t* B = Bg + bz * strideB;

  v8f acc[2][4];
  #pragma unroll
  for (int i = 0; i < 2; ++i)
    #pragma unroll
    for (int j = 0; j < 4; ++j) { v8f z = {}; acc[i][j] = z; }

  uint4 ra[2];    // staged A chunks (64x32 / 128 thr = 2 x 16B)
  uint4 rb[4];    // staged B chunks (128x32 / 128 thr = 4 x 16B), both flavors

  const int nsteps = K >> 5;

  auto loadA = [&](int k0) {
    #pragma unroll
    for (int r = 0; r < 2; ++r) {
      int ch = t + r * 128;                    // 256 chunks
      int row = ch >> 2, c8 = (ch & 3) << 3;
      ra[r] = *(const uint4*)(A + (size_t)(m0 + row) * lda + (k0 + c8));
    }
  };
  auto storeA = [&](int buf) {
    #pragma unroll
    for (int r = 0; r < 2; ++r) {
      int ch = t + r * 128;
      int row = ch >> 2, c8 = (ch & 3) << 3;
      *(uint4*)&Als[buf][row][c8] = ra[r];
    }
  };
  auto loadB = [&](int k0) {
    if (BT) {           // B is N x K row-major: tile rows are k-contiguous
      #pragma unroll
      for (int r = 0; r < 4; ++r) {
        int ch = t + r * 128;                  // 512 chunks
        int row = ch >> 2, c8 = (ch & 3) << 3;
        rb[r] = *(const uint4*)(B + (size_t)(n0 + row) * ldb + (k0 + c8));
      }
    } else {            // B is K x N row-major: vector loads of 8 along n
      #pragma unroll
      for (int r = 0; r < 4; ++r) {
        int ch = t + r * 128;                  // 512 chunks of 8 elems
        int kk = ch >> 4;                      // 0..31  (16 chunks per k-row)
        int n8 = (ch & 15) << 3;               // 0..120
        rb[r] = *(const uint4*)(B + (size_t)(k0 + kk) * ldb + (n0 + n8));
      }
    }
  };
  auto storeB = [&](int buf) {
    if (BT) {
      #pragma unroll
      for (int r = 0; r < 4; ++r) {
        int ch = t + r * 128;
        int row = ch >> 2, c8 = (ch & 3) << 3;
        *(uint4*)&Bls[buf][row][c8] = rb[r];
      }
    } else {            // transpose on LDS store: 8 b16 stores per chunk
      #pragma unroll
      for (int r = 0; r < 4; ++r) {
        int ch = t + r * 128;
        int kk = ch >> 4;
        int n8 = (ch & 15) << 3;
        bf16_t tmp[8];
        *(uint4*)tmp = rb[r];
        #pragma unroll
        for (int j = 0; j < 8; ++j)
          Bls[buf][n8 + j][kk] = tmp[j];
      }
    }
  };

  // ---- prologue: stage step 0 ----
  loadA(0); loadB(0);
  storeA(0); storeB(0);
  __syncthreads();

  for (int s = 0; s < nsteps; ++s) {
    const int cur = s & 1, nxt = cur ^ 1;
    if (s + 1 < nsteps) { loadA((s + 1) << 5); loadB((s + 1) << 5); }
    if (s + 2 < nsteps) {                       // L2 prefetch of step s+2 tiles
      const int kp = (s + 2) << 5;
      __builtin_prefetch(A + (size_t)(m0 + (t & 63)) * lda + kp + ((t >> 6) << 4), 0, 1);
      if (BT) __builtin_prefetch(B + (size_t)(n0 + (t & 127)) * ldb + kp, 0, 1);
      else    __builtin_prefetch(B + (size_t)(kp + (t & 31)) * ldb + n0 + ((t >> 5) << 5), 0, 1);
    }
    // ---- compute from lds[cur]: fragments per ISA 16-bit A/B lane layouts ----
    v16bf af[2], bfr[4];
    #pragma unroll
    for (int mi = 0; mi < 2; ++mi) {
      int row = wm + mi * 16 + l16;
      af[mi] = frag_cat(*(const v8bf*)&Als[cur][row][8 * half],
                        *(const v8bf*)&Als[cur][row][16 + 8 * half]);
    }
    #pragma unroll
    for (int ni = 0; ni < 4; ++ni) {
      int row = wn + ni * 16 + l16;
      bfr[ni] = frag_cat(*(const v8bf*)&Bls[cur][row][8 * half],
                         *(const v8bf*)&Bls[cur][row][16 + 8 * half]);
    }
    #pragma unroll
    for (int mi = 0; mi < 2; ++mi)
      #pragma unroll
      for (int ni = 0; ni < 4; ++ni)
        acc[mi][ni] = __builtin_amdgcn_wmma_f32_16x16x32_bf16(
            false, af[mi], false, bfr[ni], (short)0, acc[mi][ni], false, false);
    if (s + 1 < nsteps) { storeA(nxt); storeB(nxt); }
    __syncthreads();
  }

  // ---- epilogue: D lane layout  M = 8*half + j, N = l16 ----
  #pragma unroll
  for (int mi = 0; mi < 2; ++mi)
    #pragma unroll
    for (int ni = 0; ni < 4; ++ni)
      #pragma unroll
      for (int j = 0; j < 8; ++j) {
        int gm = m0 + wm + mi * 16 + 8 * half + j;
        int gn = n0 + wn + ni * 16 + l16;
        float v = alpha * acc[mi][ni][j];
        if (HAS_BIAS) v += bias[gn];
        if (HAS_RES)  v += res[bz * strideR + (size_t)gm * ldc + gn];
        if (RELU)     v = fmaxf(v, 0.0f);
        size_t o = bz * strideC + (size_t)gm * ldc + gn;
        if (OBF16) ((bf16_t*)Cv)[o] = (bf16_t)v;
        else       ((float*)Cv)[o]  = v;
      }
}

// ---------------------------------------------------------------------------
// Conv1d(k=7, same) as implicit-im2col WMMA GEMM. K = KW*C; each 32-wide K
// step lies inside one tap (C % 32 == 0), so the A stage is a shifted-row copy.
// Tile 64(s) x 128(cout), double-buffered like the GEMM.
// out[b,s,o] = sum_{kw,i} h[b,s+kw-3,i] * W[kw*C+i, o] + bias[o] + out[b,s,o]
// ---------------------------------------------------------------------------
__global__ __launch_bounds__(128)
void conv_wmma(const bf16_t* __restrict__ hin,  // (B,S,C) bf16
               const bf16_t* __restrict__ W,    // (KW*C, C) bf16
               const float*  __restrict__ bias, // (C)
               float* __restrict__ outp)        // (B,S,C) fp32, in-place residual
{
  __shared__ __align__(16) bf16_t Als[2][64][40];
  __shared__ __align__(16) bf16_t Bls[2][128][40];
  const int t    = threadIdx.x;
  const int lane = t & 31, wave = t >> 5;
  const int half = lane >> 4, l16 = lane & 15;
  const int wm = (wave >> 1) << 5, wn = (wave & 1) << 6;
  const int m0 = blockIdx.y * 64;      // sequence tile
  const int n0 = blockIdx.x * 128;     // cout tile
  const size_t bz = blockIdx.z;
  const bf16_t* hb = hin + bz * (size_t)S_ * C_;
  const int pad = (KW_ - 1) / 2;
  const int nsteps = (KW_ * C_) >> 5;

  v8f acc[2][4];
  #pragma unroll
  for (int i = 0; i < 2; ++i)
    #pragma unroll
    for (int j = 0; j < 4; ++j) { v8f z = {}; acc[i][j] = z; }

  uint4 ra[2];
  uint4 rb[4];

  auto loadA = [&](int k0) {
    const int kw  = k0 / C_;
    const int ci0 = k0 % C_;
    #pragma unroll
    for (int r = 0; r < 2; ++r) {
      int ch = t + r * 128;
      int row = ch >> 2, c8 = (ch & 3) << 3;
      int s = m0 + row + kw - pad;
      uint4 val = make_uint4(0u, 0u, 0u, 0u);
      if (s >= 0 && s < S_)
        val = *(const uint4*)(hb + (size_t)s * C_ + ci0 + c8);
      ra[r] = val;
    }
  };
  auto storeA = [&](int buf) {
    #pragma unroll
    for (int r = 0; r < 2; ++r) {
      int ch = t + r * 128;
      int row = ch >> 2, c8 = (ch & 3) << 3;
      *(uint4*)&Als[buf][row][c8] = ra[r];
    }
  };
  auto loadB = [&](int k0) {
    #pragma unroll
    for (int r = 0; r < 4; ++r) {
      int ch = t + r * 128;
      int kk = ch >> 4;
      int n8 = (ch & 15) << 3;
      rb[r] = *(const uint4*)(W + (size_t)(k0 + kk) * C_ + (n0 + n8));
    }
  };
  auto storeB = [&](int buf) {
    #pragma unroll
    for (int r = 0; r < 4; ++r) {
      int ch = t + r * 128;
      int kk = ch >> 4;
      int n8 = (ch & 15) << 3;
      bf16_t tmp[8];
      *(uint4*)tmp = rb[r];
      #pragma unroll
      for (int j = 0; j < 8; ++j)
        Bls[buf][n8 + j][kk] = tmp[j];
    }
  };

  loadA(0); loadB(0);
  storeA(0); storeB(0);
  __syncthreads();

  for (int s = 0; s < nsteps; ++s) {
    const int cur = s & 1, nxt = cur ^ 1;
    if (s + 1 < nsteps) { loadA((s + 1) << 5); loadB((s + 1) << 5); }
    if (s + 2 < nsteps) {
      const int kp = (s + 2) << 5;
      __builtin_prefetch(W + (size_t)(kp + (t & 31)) * C_ + n0 + ((t >> 5) << 5), 0, 1);
    }
    v16bf af[2], bfr[4];
    #pragma unroll
    for (int mi = 0; mi < 2; ++mi) {
      int row = wm + mi * 16 + l16;
      af[mi] = frag_cat(*(const v8bf*)&Als[cur][row][8 * half],
                        *(const v8bf*)&Als[cur][row][16 + 8 * half]);
    }
    #pragma unroll
    for (int ni = 0; ni < 4; ++ni) {
      int row = wn + ni * 16 + l16;
      bfr[ni] = frag_cat(*(const v8bf*)&Bls[cur][row][8 * half],
                         *(const v8bf*)&Bls[cur][row][16 + 8 * half]);
    }
    #pragma unroll
    for (int mi = 0; mi < 2; ++mi)
      #pragma unroll
      for (int ni = 0; ni < 4; ++ni)
        acc[mi][ni] = __builtin_amdgcn_wmma_f32_16x16x32_bf16(
            false, af[mi], false, bfr[ni], (short)0, acc[mi][ni], false, false);
    if (s + 1 < nsteps) { storeA(nxt); storeB(nxt); }
    __syncthreads();
  }

  #pragma unroll
  for (int mi = 0; mi < 2; ++mi)
    #pragma unroll
    for (int ni = 0; ni < 4; ++ni)
      #pragma unroll
      for (int j = 0; j < 8; ++j) {
        int gm = m0 + wm + mi * 16 + 8 * half + j;
        int gn = n0 + wn + ni * 16 + l16;
        size_t o = bz * (size_t)S_ * C_ + (size_t)gm * C_ + gn;
        outp[o] = acc[mi][ni][j] + bias[gn] + outp[o];
      }
}

// ---------------------------------------------------------------------------
// LayerNorm over C=256: one wave per row (8 rows / 256-thread block), bf16 out.
// ---------------------------------------------------------------------------
__global__ __launch_bounds__(256)
void ln_kernel(const float* __restrict__ in, const float* __restrict__ g,
               const float* __restrict__ b, bf16_t* __restrict__ out)
{
  const int wave = threadIdx.x >> 5, lane = threadIdx.x & 31;
  const size_t row = (size_t)blockIdx.x * 8 + wave;
  const float* p = in + row * C_;
  float v[8], s = 0.0f;
  #pragma unroll
  for (int i = 0; i < 8; ++i) { v[i] = p[i * 32 + lane]; s += v[i]; }
  #pragma unroll
  for (int o = 16; o > 0; o >>= 1) s += __shfl_xor(s, o, 32);
  const float mean = s * (1.0f / 256.0f);
  float s2 = 0.0f;
  #pragma unroll
  for (int i = 0; i < 8; ++i) { float d = v[i] - mean; s2 += d * d; }
  #pragma unroll
  for (int o = 16; o > 0; o >>= 1) s2 += __shfl_xor(s2, o, 32);
  const float rstd = rsqrtf(s2 * (1.0f / 256.0f) + 1e-5f);
  #pragma unroll
  for (int i = 0; i < 8; ++i) {
    int c = i * 32 + lane;
    out[row * C_ + c] = (bf16_t)((v[i] - mean) * rstd * g[c] + b[c]);
  }
}

// out = x + sinusoidal position encoding
__global__ void posenc_kernel(const float* __restrict__ x, float* __restrict__ o)
{
  size_t idx = (size_t)blockIdx.x * 256 + threadIdx.x;
  int c = (int)(idx & (C_ - 1));
  int s = (int)((idx >> 8) & (S_ - 1));
  int i = c >> 1;
  float freq = __expf((2.0f * (float)i / (float)C_) * 9.210340371976184f); // ln(1e4)
  float ang  = (float)s / freq;
  float pe   = (c & 1) ? __cosf(ang) : __sinf(ang);
  o[idx] = x[idx] + pe;
}

// Softmax over the BATCH axis (faithful to reference): in-place on bf16 scores
// laid out (H,B,S,S). One thread per (h,s,t); 32 batch values in registers.
__global__ void softmax_batch_kernel(bf16_t* __restrict__ sc)
{
  size_t idx = (size_t)blockIdx.x * 256 + threadIdx.x;  // over H*S*S
  size_t st  = idx & ((size_t)S_ * S_ - 1);
  size_t h   = idx >> 18;                                // S*S = 2^18
  size_t base = h * (size_t)B_ * S_ * S_ + st;
  const size_t stride = (size_t)S_ * S_;
  float r[B_];
  float mx = -3.0e38f;
  #pragma unroll
  for (int bb = 0; bb < B_; ++bb) {
    float f = (float)sc[base + (size_t)bb * stride];
    r[bb] = f; mx = fmaxf(mx, f);
  }
  float sum = 0.0f;
  #pragma unroll
  for (int bb = 0; bb < B_; ++bb) { r[bb] = __expf(r[bb] - mx); sum += r[bb]; }
  const float inv = 1.0f / sum;
  #pragma unroll
  for (int bb = 0; bb < B_; ++bb)
    sc[base + (size_t)bb * stride] = (bf16_t)(r[bb] * inv);
}

// cat[b,s,h*C+c] = o[h,b,s,c]
__global__ void cat_kernel(const bf16_t* __restrict__ o, bf16_t* __restrict__ cat)
{
  size_t idx = (size_t)blockIdx.x * 256 + threadIdx.x;   // over B*S*H*C
  int    c  = (int)(idx & (C_ - 1));
  int    hh = (int)((idx >> 8) & (H_ - 1));
  size_t bs = idx >> 11;                                  // b*S + s
  cat[idx] = o[((size_t)hh * B_ * S_ + bs) * C_ + c];
}

__global__ void cvt_kernel(const float* __restrict__ in, bf16_t* __restrict__ out, int n)
{
  int i = blockIdx.x * 256 + threadIdx.x;
  if (i < n) out[i] = (bf16_t)in[i];
}

// ---------------------------------------------------------------------------
extern "C" void kernel_launch(void* const* d_in, const int* in_sizes, int n_in,
                              void* d_out, int out_size, void* d_ws, size_t ws_size,
                              hipStream_t stream)
{
  (void)in_sizes; (void)n_in; (void)out_size; (void)ws_size;
  const float* x      = (const float*)d_in[0];
  const float* conv_w = (const float*)d_in[1];
  const float* conv_b = (const float*)d_in[2];
  const float* ln_g   = (const float*)d_in[3];
  const float* ln_b   = (const float*)d_in[4];
  const float* wq     = (const float*)d_in[5];
  const float* wk     = (const float*)d_in[6];
  const float* wv     = (const float*)d_in[7];
  const float* w_proj = (const float*)d_in[8];
  const float* ff_w   = (const float*)d_in[9];
  const float* ff_b   = (const float*)d_in[10];
  const float* ff2_w  = (const float*)d_in[11];
  const float* ff2_b  = (const float*)d_in[12];
  float* outF = (float*)d_out;

  char* ws = (char*)d_ws;
  size_t off = 0;
  auto take = [&](size_t bytes) -> void* {
    void* p = (void*)(ws + off);
    off += (bytes + 255) & ~(size_t)255;
    return p;
  };
  const size_t BSC = (size_t)B_ * S_ * C_;
  float*  OUT = (float*) take(BSC * 4);                        // running residual
  bf16_t* Hb  = (bf16_t*)take(BSC * 2);                        // LN output
  bf16_t* Wc  = (bf16_t*)take((size_t)NC_ * KW_ * C_ * C_ * 2);
  bf16_t* Wq  = (bf16_t*)take((size_t)H_ * C_ * C_ * 2);
  bf16_t* Wk  = (bf16_t*)take((size_t)H_ * C_ * C_ * 2);
  bf16_t* Wv  = (bf16_t*)take((size_t)H_ * C_ * C_ * 2);
  bf16_t* Wp  = (bf16_t*)take((size_t)H_ * C_ * C_ * 2);
  bf16_t* Wf1 = (bf16_t*)take((size_t)C_ * C_ * 2);
  bf16_t* Wf2 = (bf16_t*)take((size_t)C_ * C_ * 2);
  bf16_t* Qb  = (bf16_t*)take((size_t)H_ * BSC * 2);
  bf16_t* Kb  = (bf16_t*)take((size_t)H_ * BSC * 2);
  bf16_t* Vb  = (bf16_t*)take((size_t)H_ * BSC * 2);
  bf16_t* Ob  = (bf16_t*)take((size_t)H_ * BSC * 2);
  bf16_t* SC  = (bf16_t*)take((size_t)H_ * B_ * S_ * S_ * 2);  // scores/attn
  bf16_t* CAT = SC;   // reuse: scores dead after attn@V
  bf16_t* FFH = Qb;   // reuse: Q dead after scores

  auto cvt = [&](const float* s, bf16_t* d, size_t n) {
    cvt_kernel<<<dim3((unsigned)((n + 255) / 256)), dim3(256), 0, stream>>>(s, d, (int)n);
  };
  cvt(conv_w, Wc,  (size_t)NC_ * KW_ * C_ * C_);
  cvt(wq,     Wq,  (size_t)H_ * C_ * C_);
  cvt(wk,     Wk,  (size_t)H_ * C_ * C_);
  cvt(wv,     Wv,  (size_t)H_ * C_ * C_);
  cvt(w_proj, Wp,  (size_t)H_ * C_ * C_);
  cvt(ff_w,   Wf1, (size_t)C_ * C_);
  cvt(ff2_w,  Wf2, (size_t)C_ * C_);

  // positional encoding
  posenc_kernel<<<dim3((unsigned)(BSC / 256)), dim3(256), 0, stream>>>(x, OUT);

  const dim3 lnGrid((unsigned)(B_ * S_ / 8)), lnBlk(256);

  // conv stack
  for (int k = 0; k < NC_; ++k) {
    ln_kernel<<<lnGrid, lnBlk, 0, stream>>>(OUT, ln_g, ln_b, Hb);
    conv_wmma<<<dim3(C_ / 128, S_ / 64, B_), dim3(128), 0, stream>>>(
        Hb, Wc + (size_t)k * KW_ * C_ * C_, conv_b + k * C_, OUT);
  }

  // attention
  ln_kernel<<<lnGrid, lnBlk, 0, stream>>>(OUT, ln_g, ln_b, Hb);
  const int M = B_ * S_;
  // Q/K/V: batched over heads, shared A (strideA = 0)
  gemm_bf16_wmma<false, true, false, false, false>
      <<<dim3(C_ / 128, M / 64, H_), dim3(128), 0, stream>>>(
      Hb, 0, C_, Wq, (size_t)C_ * C_, C_, Qb, BSC, C_,
      nullptr, nullptr, 0, 1.0f, M, C_, C_);
  gemm_bf16_wmma<false, true, false, false, false>
      <<<dim3(C_ / 128, M / 64, H_), dim3(128), 0, stream>>>(
      Hb, 0, C_, Wk, (size_t)C_ * C_, C_, Kb, BSC, C_,
      nullptr, nullptr, 0, 0.0625f /* 1/sqrt(C) */, M, C_, C_);
  gemm_bf16_wmma<false, true, false, false, false>
      <<<dim3(C_ / 128, M / 64, H_), dim3(128), 0, stream>>>(
      Hb, 0, C_, Wv, (size_t)C_ * C_, C_, Vb, BSC, C_,
      nullptr, nullptr, 0, 1.0f, M, C_, C_);
  // scores = Q @ K^T  (NT), batched over (h,b)
  gemm_bf16_wmma<true, true, false, false, false>
      <<<dim3(S_ / 128, S_ / 64, H_ * B_), dim3(128), 0, stream>>>(
      Qb, (size_t)S_ * C_, C_, Kb, (size_t)S_ * C_, C_, SC, (size_t)S_ * S_, S_,
      nullptr, nullptr, 0, 1.0f, S_, S_, C_);
  // softmax over batch axis, in place
  softmax_batch_kernel<<<dim3((unsigned)((size_t)H_ * S_ * S_ / 256)), dim3(256), 0, stream>>>(SC);
  // o = attn @ V
  gemm_bf16_wmma<false, true, false, false, false>
      <<<dim3(C_ / 128, S_ / 64, H_ * B_), dim3(128), 0, stream>>>(
      SC, (size_t)S_ * S_, S_, Vb, (size_t)S_ * C_, C_, Ob, (size_t)S_ * C_, C_,
      nullptr, nullptr, 0, 1.0f, S_, C_, S_);
  // concat heads, then proj + residual (in-place on OUT)
  cat_kernel<<<dim3((unsigned)((size_t)B_ * S_ * H_ * C_ / 256)), dim3(256), 0, stream>>>(Ob, CAT);
  gemm_bf16_wmma<false, false, false, false, true>
      <<<dim3(C_ / 128, M / 64, 1), dim3(128), 0, stream>>>(
      CAT, 0, H_ * C_, Wp, 0, C_, OUT, 0, C_,
      nullptr, OUT, 0, 1.0f, M, C_, H_ * C_);

  // feed-forward
  ln_kernel<<<lnGrid, lnBlk, 0, stream>>>(OUT, ln_g, ln_b, Hb);
  gemm_bf16_wmma<false, true, true, true, false>
      <<<dim3(C_ / 128, M / 64, 1), dim3(128), 0, stream>>>(
      Hb, 0, C_, Wf1, 0, C_, FFH, 0, C_,
      ff_b, nullptr, 0, 1.0f, M, C_, C_);
  gemm_bf16_wmma<false, false, false, true, true>
      <<<dim3(C_ / 128, M / 64, 1), dim3(128), 0, stream>>>(
      FFH, 0, C_, Wf2, 0, C_, outF, 0, C_,
      ff2_b, OUT, 0, 1.0f, M, C_, C_);
}
